// QuantLinearMarlin_24936580120967
// MI455X (gfx1250) — matrix operand order repaired and checked by reading, not compiled, not run on hardware
//
#include <hip/hip_runtime.h>
#include <stdint.h>

typedef __attribute__((ext_vector_type(16))) _Float16 v16h;
typedef __attribute__((ext_vector_type(8)))  float    v8f;
typedef __attribute__((ext_vector_type(2)))  _Float16 h2;

static constexpr int M_DIM = 32;
static constexpr int K_DIM = 8192;
static constexpr int N_DIM = 28672;
static constexpr int KT    = K_DIM / 32;      // 256 k-tiles of depth 32
static constexpr int WAVES_PER_BLOCK = 8;     // each wave owns a 16-col N tile
static constexpr int BLOCK = 32 * WAVES_PER_BLOCK;          // 256 threads
static constexpr int NCOLS_PER_BLOCK = 16 * WAVES_PER_BLOCK; // 128 columns

union AFrag { v16h v; uint4 q[2]; };
union BFrag { v16h v; h2 p[8]; };

// [h0..h7] (8 halves in a uint4) -> pairs (h0,h4),(h1,h5),(h2,h6),(h3,h7)
// Matches the K-permutation implied by the lop3 nibble extraction below.
__device__ __forceinline__ uint4 permute8(uint4 q) {
  uint4 r;
  r.x = (q.x & 0xFFFFu) | (q.z << 16);
  r.y = (q.x >> 16)     | (q.z & 0xFFFF0000u);
  r.z = (q.y & 0xFFFFu) | (q.w << 16);
  r.w = (q.y >> 16)     | (q.w & 0xFFFF0000u);
  return r;
}

// int4x8 word -> 4 half2 values (nibble-8)*scale, K order [i, i+4] per pair.
// (1024+n)*s - 1032*s = (n-8)*s ; magic 0x6400 == fp16(1024.0)
__device__ __forceinline__ void dequant_word(uint32_t w, h2 s2, h2 c2, h2* p) {
  uint32_t u0 = ( w        & 0x000F000Fu) | 0x64006400u;
  uint32_t u1 = ((w >> 4)  & 0x000F000Fu) | 0x64006400u;
  uint32_t u2 = ((w >> 8)  & 0x000F000Fu) | 0x64006400u;
  uint32_t u3 = ((w >> 12) & 0x000F000Fu) | 0x64006400u;
  p[0] = __builtin_bit_cast(h2, u0) * s2 + c2;   // v_pk_fma_f16
  p[1] = __builtin_bit_cast(h2, u1) * s2 + c2;
  p[2] = __builtin_bit_cast(h2, u2) * s2 + c2;
  p[3] = __builtin_bit_cast(h2, u3) * s2 + c2;
}

// One-shot repack of x (32x8192 fp16, 512 KB) into WMMA-A fragment order with
// the same per-8 K permutation as the B dequant. Layout (uint4 units):
//   xs[ ((t*2 + mtile)*32 + lane)*2 + j ]
__global__ __launch_bounds__(256)
void repack_x_kernel(const _Float16* __restrict__ x, uint4* __restrict__ xs) {
  const int tid   = blockIdx.x * 256 + threadIdx.x;  // [0, KT*2*32*2)
  const int j     = tid & 1;
  const int lane  = (tid >> 1) & 31;
  const int mtile = (tid >> 6) & 1;
  const int t     = tid >> 7;
  const int row   = mtile * 16 + (lane & 15);
  const int k0    = t * 32 + (lane >> 4) * 16 + j * 8;   // 16B aligned
  const uint4 q   = *(const uint4*)(x + row * K_DIM + k0);
  xs[tid] = permute8(q);
}

template <bool REPACKED>
__global__ __launch_bounds__(BLOCK)
void w4a16_gemm_kernel(const _Float16* __restrict__ x,
                       const int*      __restrict__ qw,
                       const _Float16* __restrict__ scales,
                       const _Float16* __restrict__ bias,
                       _Float16*       __restrict__ out,
                       const uint4*    __restrict__ xs) {
  const int lane  = threadIdx.x & 31;
  const int wave  = threadIdx.x >> 5;
  const int khalf = lane >> 4;                 // which half of the K slots
  const int col   = lane & 15;
  const int colg  = blockIdx.x * NCOLS_PER_BLOCK + wave * 16 + col;

  v8f acc0 = {};   // rows 0..15
  v8f acc1 = {};   // rows 16..31

  // group (128 K) == 4 k-tiles: hoist the scale load
  for (int tg = 0; tg < KT / 4; ++tg) {
    const float sf = (float)scales[tg * N_DIM + colg];
    const _Float16 s = (_Float16)sf;
    const _Float16 c = (_Float16)(sf * -1032.0f);
    const h2 s2 = {s, s};
    const h2 c2 = {c, c};
#pragma unroll
    for (int ti = 0; ti < 4; ++ti) {
      const int t = tg * 4 + ti;
      // B: lane = column, two packed words cover this lane-half's 16 K values.
      // lanes 0-15 read row r (64B coalesced), lanes 16-31 read row r+2.
      const int r = t * 4 + khalf * 2;
      const uint32_t w0 = (uint32_t)qw[(r + 0) * N_DIM + colg];
      const uint32_t w1 = (uint32_t)qw[(r + 1) * N_DIM + colg];
      BFrag b;
      dequant_word(w0, s2, c2, &b.p[0]);
      dequant_word(w1, s2, c2, &b.p[4]);

      AFrag a0, a1;
      if (REPACKED) {
        const uint4* pa = xs + ((size_t)(t * 2 + 0) * 32 + lane) * 2;
        a0.q[0] = pa[0]; a0.q[1] = pa[1];
        const uint4* pb = xs + ((size_t)(t * 2 + 1) * 32 + lane) * 2;
        a1.q[0] = pb[0]; a1.q[1] = pb[1];
      } else {
        const int row = lane & 15;
        const int k0  = t * 32 + khalf * 16;
        a0.q[0] = permute8(*(const uint4*)(x + row * K_DIM + k0));
        a0.q[1] = permute8(*(const uint4*)(x + row * K_DIM + k0 + 8));
        a1.q[0] = permute8(*(const uint4*)(x + (row + 16) * K_DIM + k0));
        a1.q[1] = permute8(*(const uint4*)(x + (row + 16) * K_DIM + k0 + 8));
      }

      acc0 = __builtin_amdgcn_wmma_f32_16x16x32_f16(
          false, a0.v, false, b.v, (short)0, acc0, false, false);
      acc1 = __builtin_amdgcn_wmma_f32_16x16x32_f16(
          false, a1.v, false, b.v, (short)0, acc1, false, false);
    }
  }

  // Epilogue: C layout — VGPR r: lanes 0-15 -> M=r, lanes 16-31 -> M=r+8
  const float bz = (float)bias[colg];
#pragma unroll
  for (int r = 0; r < 8; ++r) {
    const int mrow = r + khalf * 8;
    out[(size_t)mrow        * N_DIM + colg] = (_Float16)(acc0[r] + bz);
    out[(size_t)(mrow + 16) * N_DIM + colg] = (_Float16)(acc1[r] + bz);
  }
}

extern "C" void kernel_launch(void* const* d_in, const int* in_sizes, int n_in,
                              void* d_out, int out_size, void* d_ws, size_t ws_size,
                              hipStream_t stream) {
  const _Float16* x      = (const _Float16*)d_in[0];
  const int*      qw     = (const int*)d_in[1];
  const _Float16* scales = (const _Float16*)d_in[2];
  const _Float16* bias   = (const _Float16*)d_in[3];
  _Float16*       out    = (_Float16*)d_out;

  const size_t xs_bytes = (size_t)M_DIM * K_DIM * sizeof(_Float16);  // 512 KB
  const dim3 grid(N_DIM / NCOLS_PER_BLOCK);                          // 224 blocks

  if (ws_size >= xs_bytes) {
    uint4* xs = (uint4*)d_ws;
    repack_x_kernel<<<(KT * 128) / 256, 256, 0, stream>>>(x, xs);
    w4a16_gemm_kernel<true><<<grid, BLOCK, 0, stream>>>(x, qw, scales, bias, out, xs);
  } else {
    w4a16_gemm_kernel<false><<<grid, BLOCK, 0, stream>>>(x, qw, scales, bias, out, nullptr);
  }
}
